// LlamaAttention_55336358642260
// MI455X (gfx1250) — compile-verified, matrix operand order checked by reading, not run on hardware
//
#include <hip/hip_runtime.h>
#include <hip/hip_bf16.h>
#include <stdint.h>

#define HIDDEN 2048
#define NH 16
#define HD 128
#define SEQ 2048
#define BATCH 2

typedef __attribute__((ext_vector_type(16))) __bf16 v16bf;
typedef __attribute__((ext_vector_type(8)))  float  v8f;

#if defined(__has_builtin)
# if __has_builtin(__builtin_amdgcn_tensor_load_to_lds)
#  define HAVE_TDM 1
# endif
#endif
#ifndef HAVE_TDM
# define HAVE_TDM 0
#endif

union Frag {
    v16bf v;
    uint4 q[2];
};

__device__ __forceinline__ uint16_t f2bf(float f) {
    uint32_t u = __float_as_uint(f);
    u += 0x7fffu + ((u >> 16) & 1u);   // round-to-nearest-even
    return (uint16_t)(u >> 16);
}

// 16x32 bf16 WMMA A-layout fragment from a row-major tile (works for global
// or LDS tiles; also serves as the B^T-fragment loader).  ISA 7.12.2:
// lane r=lane&15 holds row r; elems 0..7 = K k0..k0+7, elems 8..15 =
// K k0+16..k0+23, k0 = (lane>=16 ? 8 : 0).
__device__ __forceinline__ Frag load_frag(const uint16_t* p, int ld) {
    const int lane = threadIdx.x & 31;
    const int r = lane & 15;
    const int hi = lane >> 4;
    const uint16_t* base = p + (size_t)r * ld + hi * 8;
    Frag f;
    f.q[0] = *(const uint4*)(base);
    f.q[1] = *(const uint4*)(base + 16);
    return f;
}

__device__ __forceinline__ v8f wmma_bf16(const Frag& a, const Frag& b, v8f c) {
    return __builtin_amdgcn_wmma_f32_16x16x32_bf16(
        false, a.v, false, b.v, (short)0, c, false, false);
}

// ---------------------------------------------------------------------------
// Tile staging: TDM (tensor_load_to_lds) when available, cooperative
// load+ds_store fallback otherwise.
// ---------------------------------------------------------------------------
#if HAVE_TDM
typedef unsigned int v4u  __attribute__((ext_vector_type(4)));
typedef int          v8i_ __attribute__((ext_vector_type(8)));
typedef int          v4i_ __attribute__((ext_vector_type(4)));

// 2-D tile load: tile_cols elements (contiguous, dim0) x tile_rows rows,
// row pitch row_stride elements, bf16 (data_size=2B), dense row-major in LDS.
__device__ __forceinline__ void tdm_load_2d(const uint16_t* gsrc,
                                            uint16_t* ldst,
                                            int tile_cols, int tile_rows,
                                            int row_stride) {
    const uint64_t ga = (uint64_t)(uintptr_t)gsrc;       // byte address
    const uint32_t la = (uint32_t)(uintptr_t)ldst;       // LDS byte address
    v4u g0;
    g0[0] = 1u;                                          // count=1 (valid D#)
    g0[1] = la;                                          // lds_addr
    g0[2] = (uint32_t)ga;                                // global_addr[31:0]
    g0[3] = (uint32_t)((ga >> 32) & 0x01FFFFFFu) | (2u << 30); // [120:64],type=2
    const uint32_t td0 = (uint32_t)row_stride;           // tensor_dim0
    const uint32_t td1 = (uint32_t)tile_rows;            // tensor_dim1
    v8i_ g1;
    g1[0] = (int)(1u << 16);                             // data_size=1 -> 2B
    g1[1] = (int)((td0 & 0xFFFFu) << 16);                // dim0[15:0]
    g1[2] = (int)((td0 >> 16) | ((td1 & 0xFFFFu) << 16));// dim0 hi | dim1 lo
    g1[3] = (int)((td1 >> 16) | ((uint32_t)tile_cols << 16)); // dim1 hi|tile0
    g1[4] = (int)((uint32_t)tile_rows);                  // tile_dim1, tile2=0
    g1[5] = (int)((uint32_t)row_stride);                 // dim0_stride[31:0]
    g1[6] = 0;                                           // stride hi, dim1str
    g1[7] = 0;
    const v4i_ gz4 = {0, 0, 0, 0};
    const v8i_ gz8 = {0, 0, 0, 0, 0, 0, 0, 0};
    __builtin_amdgcn_tensor_load_to_lds(g0, g1, gz4, gz4, gz8, 0);
}
#endif

__device__ __forceinline__ void stage_tile(const uint16_t* g, int gld,
                                           uint16_t* l, int rows, int cols,
                                           int wave0) {
#if HAVE_TDM
    if (wave0) tdm_load_2d(g, l, cols, rows, gld);
#else
    (void)wave0;
    const int chunks = rows * cols / 8;                  // 16B chunks
    const int cpr = cols / 8;
    for (int ch = threadIdx.x; ch < chunks; ch += 128) {
        const int r = ch / cpr;
        const int c0 = (ch - r * cpr) * 8;
        *(uint4*)(l + r * cols + c0) =
            *(const uint4*)(g + (size_t)r * gld + c0);
    }
#endif
}

__device__ __forceinline__ void stage_wait(int wave0) {
#if HAVE_TDM
    if (wave0) {
# if defined(__has_builtin) && __has_builtin(__builtin_amdgcn_s_wait_tensorcnt)
        __builtin_amdgcn_s_wait_tensorcnt(0);
# else
        asm volatile("s_wait_tensorcnt 0x0" ::: "memory");
# endif
    }
#else
    (void)wave0;
#endif
}

// ---------------------------------------------------------------------------
// Kernel 1: fp32 -> bf16 conversion
// ---------------------------------------------------------------------------
__global__ void cvt_f32_bf16(const float* __restrict__ src,
                             uint16_t* __restrict__ dst, int n) {
    int i = blockIdx.x * blockDim.x + threadIdx.x;
    int stride = gridDim.x * blockDim.x;
    for (; i < n; i += stride) dst[i] = f2bf(src[i]);
}

// ---------------------------------------------------------------------------
// Kernel 2: QKV projection (x @ W.T) + RoPE.  Block = 4 waves sharing one
// head's weight tile (staged 128x32 per k-step via TDM, double buffered);
// each wave owns 16 M-rows.  Wave tile 16x128 = full head so RoPE pairing
// (d, d+64) is register-local (acc tile t pairs with t+4).
// Q,K stored [B,nh,S,hd] bf16 post-RoPE; V stored transposed [B,nh,hd,S].
// ---------------------------------------------------------------------------
__global__ __launch_bounds__(128) void qkv_rope_kernel(
    const uint16_t* __restrict__ Xb,
    const uint16_t* __restrict__ Wq,
    const uint16_t* __restrict__ Wk,
    const uint16_t* __restrict__ Wv,
    uint16_t* __restrict__ Qo,
    uint16_t* __restrict__ Ko,
    uint16_t* __restrict__ Vt) {
    __shared__ uint16_t bsh[2][128 * 32];                // 16 KB, double buf

    const int proj = blockIdx.y;                         // 0=Q 1=K 2=V
    const uint16_t* W = (proj == 0) ? Wq : (proj == 1) ? Wk : Wv;

    const int bx = blockIdx.x;                           // 0..1023
    const int head = bx & 15;
    const int mt = bx >> 4;                              // 64-row block tile
    const int wv = threadIdx.x >> 5;
    const int wave0 = (wv == 0);
    const int m0 = mt * 64 + wv * 16;
    const int n0 = head * HD;
    const int b = m0 / SEQ;
    const int s0 = m0 - b * SEQ;

    const uint16_t* A  = Xb + (size_t)m0 * HIDDEN;
    const uint16_t* Bt = W  + (size_t)n0 * HIDDEN;       // W natural == B^T

    stage_tile(Bt, HIDDEN, &bsh[0][0], 128, 32, wave0);

    v8f acc[8] = {};
    for (int i = 0; i < 64; ++i) {
        const int k0 = i * 32;
        stage_wait(wave0);
        __syncthreads();
        if (i + 1 < 64)
            stage_tile(Bt + k0 + 32, HIDDEN, &bsh[(i + 1) & 1][0], 128, 32,
                       wave0);
        const uint16_t* bs = &bsh[i & 1][0];
        Frag a = load_frag(A + k0, HIDDEN);
#pragma unroll
        for (int t = 0; t < 8; ++t) {
            Frag bf = load_frag(bs + t * 16 * 32, 32);
            acc[t] = wmma_bf16(a, bf, acc[t]);
        }
    }

    const int lane = threadIdx.x & 31;
    const int col = lane & 15;
    const int hi = lane >> 4;

    if (proj == 2) {
        // V^T: lane holds fixed d, 8 consecutive s rows -> one b128 store
#pragma unroll
        for (int t = 0; t < 8; ++t) {
            const int d = t * 16 + col;
            uint16_t tmp[8];
#pragma unroll
            for (int e = 0; e < 8; ++e) tmp[e] = f2bf(acc[t][e]);
            const size_t off =
                ((size_t)(b * NH + head) * HD + d) * SEQ + s0 + hi * 8;
            *(uint4*)(Vt + off) = *(const uint4*)tmp;
        }
    } else {
        uint16_t* Out = (proj == 0) ? Qo : Ko;
#pragma unroll
        for (int t = 0; t < 4; ++t) {
            const int j = t * 16 + col;                  // 0..63
            const float invf =
                exp2f(-(float)j * (13.287712379549449f / 64.0f));
#pragma unroll
            for (int e = 0; e < 8; ++e) {
                const int s = s0 + e + hi * 8;
                float sn, cs;
                __sincosf((float)s * invf, &sn, &cs);
                const float x = acc[t][e];
                const float y = acc[t + 4][e];
                const size_t base = ((size_t)(b * NH + head) * SEQ + s) * HD;
                Out[base + j]      = f2bf(x * cs - y * sn);
                Out[base + j + 64] = f2bf(y * cs + x * sn);
            }
        }
    }
}

// ---------------------------------------------------------------------------
// Kernel 3: flash attention.  Block = 4 waves on 4 consecutive q-tiles of
// ONE head, sharing K (32x128) and V^T (128x32) tiles staged per 32-key
// block via TDM (double buffered).  Per wave: 8 score WMMAs + online
// softmax (width-16 shfl row reductions) + C->A transpose of P through a
// per-wave LDS slab + 8 PV WMMAs.
// ---------------------------------------------------------------------------
__global__ __launch_bounds__(128) void attn_kernel(
    const uint16_t* __restrict__ Q,
    const uint16_t* __restrict__ Kk,
    const uint16_t* __restrict__ Vt,
    uint16_t* __restrict__ Ctx) {
    __shared__ uint16_t ksh[2][32 * 128];                // 16 KB
    __shared__ uint16_t vsh[2][128 * 32];                // 16 KB
    __shared__ uint16_t psh[4][16 * 32];                 // 4 KB per-wave P

    const int bx = blockIdx.x;                           // 0..1023
    const int hidx = bx >> 5;                            // (b*NH+h), 0..31
    const int wv = threadIdx.x >> 5;
    const int wave0 = (wv == 0);
    const int qt = (bx & 31) * 4 + wv;                   // q-tile, 0..127
    const int lane = threadIdx.x & 31;
    const int col = lane & 15;
    const int hi = lane >> 4;

    const uint16_t* Qb = Q  + ((size_t)hidx * SEQ + qt * 16) * HD;
    const uint16_t* Kb = Kk + (size_t)hidx * SEQ * HD;
    const uint16_t* Vb = Vt + (size_t)hidx * HD * SEQ;

    Frag qa[4];
#pragma unroll
    for (int i = 0; i < 4; ++i) qa[i] = load_frag(Qb + i * 32, HD);

    v8f o[8] = {};
    float mrow[8], lsum[8];
#pragma unroll
    for (int e = 0; e < 8; ++e) { mrow[e] = -3.0e38f; lsum[e] = 0.0f; }

    const float c = 0.12753102f;                         // log2(e)/sqrt(128)

    stage_tile(Kb, HD, &ksh[0][0], 32, 128, wave0);
    stage_tile(Vb, SEQ, &vsh[0][0], 128, 32, wave0);

    for (int i = 0; i < SEQ / 32; ++i) {
        const int kb = i * 32;
        stage_wait(wave0);
        __syncthreads();
        if (kb + 32 < SEQ) {
            stage_tile(Kb + (size_t)(kb + 32) * HD, HD,
                       &ksh[(i + 1) & 1][0], 32, 128, wave0);
            stage_tile(Vb + kb + 32, SEQ,
                       &vsh[(i + 1) & 1][0], 128, 32, wave0);
        }
        const uint16_t* ks = &ksh[i & 1][0];
        const uint16_t* vs = &vsh[i & 1][0];

        v8f sa = {}, sb = {};
#pragma unroll
        for (int j = 0; j < 4; ++j) {
            Frag k0 = load_frag(ks + j * 32, 128);
            sa = wmma_bf16(qa[j], k0, sa);
            Frag k1 = load_frag(ks + 16 * 128 + j * 32, 128);
            sb = wmma_bf16(qa[j], k1, sb);
        }
        // online softmax; row M = e + hi*8, columns striped across 16 lanes
#pragma unroll
        for (int e = 0; e < 8; ++e) {
            float v0 = sa[e] * c;
            float v1 = sb[e] * c;
            float mx = fmaxf(v0, v1);
#pragma unroll
            for (int off = 8; off > 0; off >>= 1)
                mx = fmaxf(mx, __shfl_xor(mx, off, 16));
            const float mn = fmaxf(mrow[e], mx);
            const float corr = exp2f(mrow[e] - mn);
            mrow[e] = mn;
            const float p0 = exp2f(v0 - mn);
            const float p1 = exp2f(v1 - mn);
            float ps = p0 + p1;
#pragma unroll
            for (int off = 8; off > 0; off >>= 1)
                ps += __shfl_xor(ps, off, 16);
            lsum[e] = lsum[e] * corr + ps;
#pragma unroll
            for (int t = 0; t < 8; ++t) o[t][e] *= corr;
            const int row = e + hi * 8;
            psh[wv][row * 32 + col]      = f2bf(p0);
            psh[wv][row * 32 + 16 + col] = f2bf(p1);
        }
        asm volatile("s_wait_dscnt 0" ::: "memory");     // intra-wave P ready
        Frag pf = load_frag(&psh[wv][0], 32);            // P in A layout
#pragma unroll
        for (int t = 0; t < 8; ++t) {
            Frag vf = load_frag(vs + t * 16 * 32, 32);
            o[t] = wmma_bf16(pf, vf, o[t]);
        }
    }

    const int b = hidx >> 4, h = hidx & 15;
#pragma unroll
    for (int e = 0; e < 8; ++e) {
        const float inv = 1.0f / lsum[e];
        const int s = qt * 16 + e + hi * 8;
        const size_t base = ((size_t)b * SEQ + s) * HIDDEN + h * HD;
#pragma unroll
        for (int t = 0; t < 8; ++t)
            Ctx[base + t * 16 + col] = f2bf(o[t][e] * inv);
    }
}

// ---------------------------------------------------------------------------
// Kernel 4: output projection ctx @ Wo.T -> fp32 (same blocked structure)
// ---------------------------------------------------------------------------
__global__ __launch_bounds__(128) void oproj_kernel(
    const uint16_t* __restrict__ Ctx,
    const uint16_t* __restrict__ Wo,
    float* __restrict__ Out) {
    __shared__ uint16_t bsh[2][128 * 32];

    const int bx = blockIdx.x;
    const int ntile = bx & 15;
    const int mt = bx >> 4;
    const int wv = threadIdx.x >> 5;
    const int wave0 = (wv == 0);
    const int m0 = mt * 64 + wv * 16;
    const int n0 = ntile * 128;

    const uint16_t* A  = Ctx + (size_t)m0 * HIDDEN;
    const uint16_t* Bt = Wo  + (size_t)n0 * HIDDEN;

    stage_tile(Bt, HIDDEN, &bsh[0][0], 128, 32, wave0);

    v8f acc[8] = {};
    for (int i = 0; i < 64; ++i) {
        const int k0 = i * 32;
        stage_wait(wave0);
        __syncthreads();
        if (i + 1 < 64)
            stage_tile(Bt + k0 + 32, HIDDEN, &bsh[(i + 1) & 1][0], 128, 32,
                       wave0);
        const uint16_t* bs = &bsh[i & 1][0];
        Frag a = load_frag(A + k0, HIDDEN);
#pragma unroll
        for (int t = 0; t < 8; ++t) {
            Frag bf = load_frag(bs + t * 16 * 32, 32);
            acc[t] = wmma_bf16(a, bf, acc[t]);
        }
    }

    const int lane = threadIdx.x & 31;
    const int col = lane & 15;
    const int hi = lane >> 4;
#pragma unroll
    for (int e = 0; e < 8; ++e) {
        const size_t base = (size_t)(m0 + e + hi * 8) * HIDDEN + n0;
#pragma unroll
        for (int t = 0; t < 8; ++t)
            Out[base + t * 16 + col] = acc[t][e];
    }
}

// ---------------------------------------------------------------------------
extern "C" void kernel_launch(void* const* d_in, const int* in_sizes, int n_in,
                              void* d_out, int out_size, void* d_ws,
                              size_t ws_size, hipStream_t stream) {
    const float* X  = (const float*)d_in[0];
    const float* Wq = (const float*)d_in[1];
    const float* Wk = (const float*)d_in[2];
    const float* Wv = (const float*)d_in[3];
    const float* Wo = (const float*)d_in[4];
    float* Out = (float*)d_out;

    const size_t XN = (size_t)BATCH * SEQ * HIDDEN;      // 8,388,608
    const size_t WN = (size_t)HIDDEN * HIDDEN;           // 4,194,304

    uint16_t* ws  = (uint16_t*)d_ws;
    uint16_t* Xb  = ws;
    uint16_t* Wqb = Xb + XN;
    uint16_t* Wkb = Wqb + WN;
    uint16_t* Wvb = Wkb + WN;
    uint16_t* Wob = Wvb + WN;
    uint16_t* Qb  = Wob + WN;                            // [B,nh,S,hd]
    uint16_t* Kb  = Qb + XN;                             // [B,nh,S,hd]
    uint16_t* Vtb = Kb + XN;                             // [B,nh,hd,S]
    uint16_t* Ctx = Vtb + XN;                            // [B,S,H]

    cvt_f32_bf16<<<2048, 256, 0, stream>>>(X,  Xb,  (int)XN);
    cvt_f32_bf16<<<1024, 256, 0, stream>>>(Wq, Wqb, (int)WN);
    cvt_f32_bf16<<<1024, 256, 0, stream>>>(Wk, Wkb, (int)WN);
    cvt_f32_bf16<<<1024, 256, 0, stream>>>(Wv, Wvb, (int)WN);
    cvt_f32_bf16<<<1024, 256, 0, stream>>>(Wo, Wob, (int)WN);

    qkv_rope_kernel<<<dim3(1024, 3), 128, 0, stream>>>(Xb, Wqb, Wkb, Wvb,
                                                       Qb, Kb, Vtb);
    attn_kernel<<<1024, 128, 0, stream>>>(Qb, Kb, Vtb, Ctx);
    oproj_kernel<<<1024, 128, 0, stream>>>(Ctx, Wob, Out);
}
